// GeometryAwareSelfAttentionBlock_7232724927304
// MI455X (gfx1250) — matrix operand order repaired and checked by reading, not compiled, not run on hardware
//
#include <hip/hip_runtime.h>

// Problem constants (fixed by reference)
#define BQ 8
#define NP 2048
#define DM 384
#define NH 6
#define DH 64
#define KNN 8
#define MROWS (BQ * NP)   // 16384

typedef __attribute__((ext_vector_type(16))) __bf16 v16bf;
typedef __attribute__((ext_vector_type(8)))  float  v8f;

union Frag16 { v16bf v; unsigned short u[16]; };

__device__ inline unsigned short f2b(float f) {
  union { float f; unsigned int u; } c; c.f = f;
  unsigned int u = c.u;
  u = (u + 0x7fffu + ((u >> 16) & 1u)) >> 16;
  return (unsigned short)u;
}
__device__ inline float b2f(unsigned short h) {
  union { unsigned int u; float f; } c; c.u = ((unsigned int)h) << 16; return c.f;
}
__device__ inline float gelu_tanh(float x) {
  const float c0 = 0.7978845608028654f, c1 = 0.044715f;
  float t = tanhf(c0 * (x + c1 * x * x * x));
  return 0.5f * x * (1.0f + t);
}

// ---- gfx1250 async global->LDS staging (guarded; falls back to VGPR path) --
#if __has_builtin(__builtin_amdgcn_global_load_async_to_lds_b128)
#define HAVE_ASYNC_LDS 1
#else
#define HAVE_ASYNC_LDS 0
#endif

typedef int vint4 __attribute__((vector_size(4 * sizeof(int))));
typedef __attribute__((address_space(1))) vint4 g_vint4;
typedef __attribute__((address_space(3))) vint4 l_vint4;
__device__ inline g_vint4* to_gbl_v4(const void* p) {
  return (g_vint4*)(unsigned long long)p;
}
__device__ inline l_vint4* to_lds_v4(void* p) {
  return (l_vint4*)(unsigned int)(unsigned long long)p;
}
__device__ inline void wait_async_lds() {
#if __has_builtin(__builtin_amdgcn_s_wait_asynccnt)
  __builtin_amdgcn_s_wait_asynccnt(0);
#else
  asm volatile("s_wait_asynccnt 0" ::: "memory");
#endif
}

// ---------------------------------------------------------------------------
// weight converts
// ---------------------------------------------------------------------------
__global__ void cvt_kernel(const float* __restrict__ src, unsigned short* __restrict__ dst, int n) {
  int i = blockIdx.x * 256 + threadIdx.x;
  if (i < n) dst[i] = f2b(src[i]);
}
// dst[c][r] = src[r][c]
__global__ void cvt_t_kernel(const float* __restrict__ src, unsigned short* __restrict__ dst,
                             int rows, int cols) {
  int i = blockIdx.x * 256 + threadIdx.x;
  if (i < rows * cols) {
    int r = i / cols, c = i % cols;
    dst[(size_t)c * rows + r] = f2b(src[i]);
  }
}

// ---------------------------------------------------------------------------
// LN1: features [B,D,N] -> f [B*N, D] f32  and  nf bf16
// ---------------------------------------------------------------------------
__global__ __launch_bounds__(128) void ln1_kernel(const float* __restrict__ feat,
                                                  const float* __restrict__ gma,
                                                  const float* __restrict__ bta,
                                                  float* __restrict__ f,
                                                  unsigned short* __restrict__ nf) {
  int n = blockIdx.x, b = blockIdx.y, t = threadIdx.x;
  __shared__ float r1[128], r2[128];
  float x[3];
  float s = 0.f, s2 = 0.f;
  for (int i = 0; i < 3; i++) {
    int d = t + i * 128;
    float v = feat[((size_t)(b * DM + d)) * NP + n];
    x[i] = v; s += v; s2 += v * v;
  }
  r1[t] = s; r2[t] = s2; __syncthreads();
  for (int w = 64; w > 0; w >>= 1) {
    if (t < w) { r1[t] += r1[t + w]; r2[t] += r2[t + w]; }
    __syncthreads();
  }
  float mean = r1[0] * (1.0f / DM);
  float var = r2[0] * (1.0f / DM) - mean * mean;
  float rstd = rsqrtf(var + 1e-5f);
  size_t row = (size_t)b * NP + n;
  for (int i = 0; i < 3; i++) {
    int d = t + i * 128;
    f[row * DM + d] = x[i];
    nf[row * DM + d] = f2b((x[i] - mean) * rstd * gma[d] + bta[d]);
  }
}

// ---------------------------------------------------------------------------
// bf16 WMMA GEMM: C[M,N] = A[M,K] @ B[K,N]; compile-time epilogue
// ---------------------------------------------------------------------------
#define F_GELU 1
#define F_BF16 2
#define F_RESID 4
#define F_BIAS 8
#define TBM 128
#define TBN 128
#define TBK 32

template <int FLAGS>
__global__ __launch_bounds__(256) void gemm_bf16_kernel(
    const unsigned short* __restrict__ A, int lda,
    const unsigned short* __restrict__ Bw, int ldb,
    const float* __restrict__ bias,
    const float* __restrict__ resid,
    float* __restrict__ outF,
    unsigned short* __restrict__ outH,
    int ldc, int N, int K) {
  __shared__ alignas(16) unsigned short As[TBM * TBK];   // 8 KB
  __shared__ alignas(16) unsigned short Bs[TBK * TBN];   // 8 KB
  int tid = threadIdx.x;
  int lane = tid & 31, wave = tid >> 5;
  int wr = wave >> 2, wc = wave & 3;          // 2 x 4 waves; each wave 64x32
  int g = lane >> 4, hl = lane & 15;
  int blockRow = blockIdx.y * TBM;
  int blockCol = blockIdx.x * TBN;

  v8f z8 = {0.f, 0.f, 0.f, 0.f, 0.f, 0.f, 0.f, 0.f};
  v8f acc[4][2];
  #pragma unroll
  for (int i = 0; i < 4; i++) { acc[i][0] = z8; acc[i][1] = z8; }

#if !HAVE_ASYNC_LDS
  unsigned int* As32 = reinterpret_cast<unsigned int*>(As);
  unsigned int* Bs32 = reinterpret_cast<unsigned int*>(Bs);
#endif

  for (int k0 = 0; k0 < K; k0 += TBK) {
#if HAVE_ASYNC_LDS
    // A tile: 128x32 ushort = 512 x 16B chunks (4 per row), 2 per thread
    #pragma unroll
    for (int i = 0; i < 2; i++) {
      int p = tid + 256 * i;
      int r = p >> 2, cp = p & 3;
      const void* gp = (const void*)(A + (size_t)(blockRow + r) * lda + k0 + cp * 8);
      __builtin_amdgcn_global_load_async_to_lds_b128(to_gbl_v4(gp), to_lds_v4(As + p * 8), 0, 0);
    }
    // B tile: 32x128 ushort = 512 x 16B chunks (16 per row), 2 per thread
    #pragma unroll
    for (int i = 0; i < 2; i++) {
      int p = tid + 256 * i;
      int r = p >> 4, cp = p & 15;
      const void* gp = (const void*)(Bw + (size_t)(k0 + r) * ldb + blockCol + cp * 8);
      __builtin_amdgcn_global_load_async_to_lds_b128(to_gbl_v4(gp), to_lds_v4(Bs + p * 8), 0, 0);
    }
    wait_async_lds();
#else
    #pragma unroll
    for (int i = 0; i < 8; i++) {
      int p = tid + 256 * i;
      int r = p >> 4, cp = p & 15;
      const unsigned int* Ag =
          reinterpret_cast<const unsigned int*>(A + (size_t)(blockRow + r) * lda + k0);
      As32[p] = Ag[cp];
    }
    #pragma unroll
    for (int i = 0; i < 8; i++) {
      int p = tid + 256 * i;
      int r = p >> 6, cp = p & 63;
      const unsigned int* Bg =
          reinterpret_cast<const unsigned int*>(Bw + (size_t)(k0 + r) * ldb + blockCol);
      Bs32[p] = Bg[cp];
    }
#endif
    __syncthreads();

    Frag16 af[4], bfr[2];
    #pragma unroll
    for (int s = 0; s < 4; s++) {
      int mrow = wr * 64 + s * 16 + hl;
      #pragma unroll
      for (int e = 0; e < 16; e++) {
        int k = e + 8 * g + 8 * (e >> 3);        // ISA A-matrix 16x32 bf16 layout
        af[s].u[e] = As[mrow * TBK + k];
      }
    }
    #pragma unroll
    for (int s = 0; s < 2; s++) {
      int ncol = wc * 32 + s * 16 + hl;
      #pragma unroll
      for (int e = 0; e < 16; e++) {
        int k = e + 8 * g + 8 * (e >> 3);        // ISA B-matrix 32x16 bf16 layout (table order, matches A)
        bfr[s].u[e] = Bs[k * TBN + ncol];
      }
    }
    #pragma unroll
    for (int ar = 0; ar < 4; ar++)
      #pragma unroll
      for (int bc = 0; bc < 2; bc++)
        acc[ar][bc] = __builtin_amdgcn_wmma_f32_16x16x32_bf16(
            false, af[ar].v, false, bfr[bc].v, (short)0, acc[ar][bc], false, false);
    __syncthreads();
  }

  #pragma unroll
  for (int ar = 0; ar < 4; ar++)
    #pragma unroll
    for (int bc = 0; bc < 2; bc++) {
      int rowBase = blockRow + wr * 64 + ar * 16 + 8 * g;
      int col = blockCol + wc * 32 + bc * 16 + hl;
      float bv = 0.0f;
      if constexpr (FLAGS & F_BIAS) bv = bias[col];
      #pragma unroll
      for (int r = 0; r < 8; r++) {
        int row = rowBase + r;
        float v = acc[ar][bc][r] + bv;
        if constexpr (FLAGS & F_GELU) v = gelu_tanh(v);
        if constexpr (FLAGS & F_RESID) {
          int bb = row >> 11, np = row & (NP - 1);
          outF[((size_t)(bb * DM + col)) * NP + np] = resid[(size_t)row * DM + col] + v;
        } else if constexpr (FLAGS & F_BF16) {
          outH[(size_t)row * ldc + col] = f2b(v);
        } else {
          outF[(size_t)row * ldc + col] = v;
        }
      }
    }
}

template <int FLAGS>
static void launch_gemm(hipStream_t stream, const unsigned short* A, int lda,
                        const unsigned short* Bw, int ldb, const float* bias,
                        const float* resid, float* oF, unsigned short* oH,
                        int ldc, int N, int K) {
  gemm_bf16_kernel<FLAGS><<<dim3(N / TBN, MROWS / TBM), 256, 0, stream>>>(
      A, lda, Bw, ldb, bias, resid, oF, oH, ldc, N, K);
}

// ---------------------------------------------------------------------------
// Flash attention: one wave per (b, h, 16 query rows). qkv bf16 [B*N, 1152].
// Online softmax; P via LDS (C-layout -> A-layout); V staged via LDS.
// ---------------------------------------------------------------------------
__global__ __launch_bounds__(32) void attn_kernel(const unsigned short* __restrict__ qkv,
                                                  unsigned short* __restrict__ attnh) {
  int qb = blockIdx.x, h = blockIdx.y, b = blockIdx.z;
  int lane = threadIdx.x;
  int g = lane >> 4, hl = lane & 15;
  const unsigned short* base = qkv + (size_t)b * NP * (3 * DM);

  Frag16 qf[2];
  {
    int qrow = qb * 16 + hl;
    const unsigned short* qp = base + (size_t)qrow * (3 * DM) + h * DH;
    #pragma unroll
    for (int fi = 0; fi < 2; fi++)
      #pragma unroll
      for (int e = 0; e < 16; e++) {
        int k = e + 8 * g + 8 * (e >> 3);
        qf[fi].u[e] = qp[fi * 32 + k];
      }
  }

  v8f z8 = {0.f, 0.f, 0.f, 0.f, 0.f, 0.f, 0.f, 0.f};
  v8f o[4]; o[0] = z8; o[1] = z8; o[2] = z8; o[3] = z8;
  float mi[8], li[8];
  #pragma unroll
  for (int r = 0; r < 8; r++) { mi[r] = -1e30f; li[r] = 0.f; }

  __shared__ alignas(16) unsigned short Pl[16 * 32];
  __shared__ alignas(16) unsigned short Vs[32 * 64];

  for (int jb = 0; jb < NP; jb += 32) {
    // stage V panel (32 keys x first 32 of 64 dims) with coalesced 16B loads
    {
      const uint4* src =
          reinterpret_cast<const uint4*>(base + (size_t)(jb + lane) * (3 * DM) + 2 * DM + h * DH);
      uint4* dst = reinterpret_cast<uint4*>(Vs + lane * 64);
      #pragma unroll
      for (int i = 0; i < 4; i++) dst[i] = src[i];
    }

    v8f s[2];
    #pragma unroll
    for (int tk = 0; tk < 2; tk++) {
      v8f accS = z8;
      int key = jb + tk * 16 + hl;
      #pragma unroll
      for (int fi = 0; fi < 2; fi++) {
        Frag16 kf;
        const unsigned short* kp = base + (size_t)key * (3 * DM) + DM + h * DH + fi * 32;
        #pragma unroll
        for (int e = 0; e < 16; e++) kf.u[e] = kp[e + 8 * g + 8 * (e >> 3)];
        accS = __builtin_amdgcn_wmma_f32_16x16x32_bf16(
            false, qf[fi].v, false, kf.v, (short)0, accS, false, false);
      }
      s[tk] = accS;
    }

    float scl[8];
    #pragma unroll
    for (int r = 0; r < 8; r++) {
      float a = s[0][r] * 0.125f, c = s[1][r] * 0.125f;     // 1/sqrt(64)
      float mrow = fmaxf(a, c);
      #pragma unroll
      for (int d = 1; d < 16; d <<= 1) mrow = fmaxf(mrow, __shfl_xor(mrow, d, 32));
      float nm = fmaxf(mi[r], mrow);
      scl[r] = __expf(mi[r] - nm);
      float e0 = __expf(a - nm), e1 = __expf(c - nm);
      s[0][r] = e0; s[1][r] = e1;
      float ps = e0 + e1;
      #pragma unroll
      for (int d = 1; d < 16; d <<= 1) ps += __shfl_xor(ps, d, 32);
      li[r] = li[r] * scl[r] + ps;
      mi[r] = nm;
    }
    #pragma unroll
    for (int t = 0; t < 4; t++)
      #pragma unroll
      for (int r = 0; r < 8; r++) o[t][r] *= scl[r];

    __syncthreads();
    #pragma unroll
    for (int tk = 0; tk < 2; tk++)
      #pragma unroll
      for (int r = 0; r < 8; r++)
        Pl[(r + 8 * g) * 32 + tk * 16 + hl] = f2b(s[tk][r]);
    __syncthreads();

    Frag16 pf;
    #pragma unroll
    for (int e = 0; e < 16; e++) {
      int k = e + 8 * g + 8 * (e >> 3);
      pf.u[e] = Pl[hl * 32 + k];
    }
    #pragma unroll
    for (int t = 0; t < 4; t++) {
      Frag16 vf;
      #pragma unroll
      for (int e = 0; e < 16; e++) {
        int kk = e + 8 * g + 8 * (e >> 3);
        vf.u[e] = Vs[kk * 64 + t * 16 + hl];
      }
      o[t] = __builtin_amdgcn_wmma_f32_16x16x32_bf16(
          false, pf.v, false, vf.v, (short)0, o[t], false, false);
    }
    __syncthreads();
  }

  #pragma unroll
  for (int r = 0; r < 8; r++) {
    float inv = 1.0f / li[r];
    int row = qb * 16 + r + 8 * g;
    #pragma unroll
    for (int t = 0; t < 4; t++)
      attnh[((size_t)(b * NP + row)) * DM + h * DH + t * 16 + hl] = f2b(o[t][r] * inv);
  }
}

// ---------------------------------------------------------------------------
// KNN top-8 (one wave per point); reference's sqn+sqj-2*dot with index ties
// ---------------------------------------------------------------------------
__global__ __launch_bounds__(32) void knn_kernel(const float* __restrict__ coords,
                                                 int* __restrict__ idx) {
  int n = blockIdx.x, b = blockIdx.y, lane = threadIdx.x;
  const float* cx = coords + (size_t)b * 3 * NP;
  float qx = cx[n], qy = cx[NP + n], qz = cx[2 * NP + n];
  float sqn = qx * qx + qy * qy + qz * qz;
  float d8[KNN]; int i8[KNN];
  #pragma unroll
  for (int s = 0; s < KNN; s++) { d8[s] = 3.0e38f; i8[s] = 0x7fffffff; }
  for (int j = lane; j < NP; j += 32) {
    float xj = cx[j], yj = cx[NP + j], zj = cx[2 * NP + j];
    float sqj = xj * xj + yj * yj + zj * zj;
    float dot = qx * xj + qy * yj + qz * zj;
    float d2 = sqn + sqj - 2.0f * dot;
    if (d2 < d8[KNN - 1] || (d2 == d8[KNN - 1] && j < i8[KNN - 1])) {
      d8[KNN - 1] = d2; i8[KNN - 1] = j;
      #pragma unroll
      for (int s = KNN - 1; s > 0; s--) {
        bool sw = (d8[s] < d8[s - 1]) || (d8[s] == d8[s - 1] && i8[s] < i8[s - 1]);
        if (!sw) break;
        float td = d8[s]; d8[s] = d8[s - 1]; d8[s - 1] = td;
        int ti = i8[s]; i8[s] = i8[s - 1]; i8[s - 1] = ti;
      }
    }
  }
  __shared__ float ld[32 * KNN];
  __shared__ int li[32 * KNN];
  #pragma unroll
  for (int s = 0; s < KNN; s++) { ld[lane * KNN + s] = d8[s]; li[lane * KNN + s] = i8[s]; }
  __syncthreads();
  if (lane == 0) {
    for (int s = 0; s < KNN; s++) {
      int best = 0; float bd = 3.0e38f; int bi = 0x7fffffff;
      for (int t = 0; t < 32 * KNN; t++) {
        if (ld[t] < bd || (ld[t] == bd && li[t] < bi)) { bd = ld[t]; bi = li[t]; best = t; }
      }
      idx[((size_t)(b * NP + n)) * KNN + s] = bi;
      ld[best] = 3.0e38f; li[best] = 0x7fffffff;
    }
  }
}

// ---------------------------------------------------------------------------
// Graph-attn aggregate: gs_k = nbr_k . u / sqrt(D); m = softmax(gs) @ nbrs
// ---------------------------------------------------------------------------
__global__ __launch_bounds__(128) void geo_kernel(const unsigned short* __restrict__ nf,
                                                  const unsigned short* __restrict__ u,
                                                  const int* __restrict__ idx,
                                                  unsigned short* __restrict__ mout) {
  int row = blockIdx.x;
  int b = row >> 11;
  int t = threadIdx.x;
  __shared__ float gs[KNN];
  __shared__ float red[128];
  int nb[KNN];
  #pragma unroll
  for (int k = 0; k < KNN; k++) nb[k] = idx[(size_t)row * KNN + k];
  const float scale = rsqrtf((float)DM);
  for (int k = 0; k < KNN; k++) {
    float p = 0.f;
    for (int i = 0; i < 3; i++) {
      int d = t + i * 128;
      p += b2f(nf[((size_t)(b * NP + nb[k])) * DM + d]) * b2f(u[(size_t)row * DM + d]);
    }
    red[t] = p; __syncthreads();
    for (int w = 64; w > 0; w >>= 1) { if (t < w) red[t] += red[t + w]; __syncthreads(); }
    if (t == 0) gs[k] = red[0] * scale;
    __syncthreads();
  }
  float mx = -1e30f;
  #pragma unroll
  for (int k = 0; k < KNN; k++) mx = fmaxf(mx, gs[k]);
  float e[KNN], se = 0.f;
  #pragma unroll
  for (int k = 0; k < KNN; k++) { e[k] = __expf(gs[k] - mx); se += e[k]; }
  float inv = 1.0f / se;
  for (int i = 0; i < 3; i++) {
    int d = t + i * 128;
    float acc = 0.f;
    #pragma unroll
    for (int k = 0; k < KNN; k++)
      acc += e[k] * inv * b2f(nf[((size_t)(b * NP + nb[k])) * DM + d]);
    mout[(size_t)row * DM + d] = f2b(acc);
  }
}

// ---------------------------------------------------------------------------
// f2 = f + merged; LN2 -> h bf16
// ---------------------------------------------------------------------------
__global__ __launch_bounds__(128) void ln2_kernel(const float* __restrict__ f,
                                                  const float* __restrict__ merged,
                                                  const float* __restrict__ gma,
                                                  const float* __restrict__ bta,
                                                  float* __restrict__ f2,
                                                  unsigned short* __restrict__ h) {
  int row = blockIdx.x, t = threadIdx.x;
  __shared__ float r1[128], r2[128];
  float x[3], s = 0.f, s2 = 0.f;
  for (int i = 0; i < 3; i++) {
    int d = t + i * 128;
    float v = f[(size_t)row * DM + d] + merged[(size_t)row * DM + d];
    x[i] = v; s += v; s2 += v * v;
  }
  r1[t] = s; r2[t] = s2; __syncthreads();
  for (int w = 64; w > 0; w >>= 1) {
    if (t < w) { r1[t] += r1[t + w]; r2[t] += r2[t + w]; }
    __syncthreads();
  }
  float mean = r1[0] * (1.0f / DM);
  float var = r2[0] * (1.0f / DM) - mean * mean;
  float rstd = rsqrtf(var + 1e-5f);
  for (int i = 0; i < 3; i++) {
    int d = t + i * 128;
    f2[(size_t)row * DM + d] = x[i];
    h[(size_t)row * DM + d] = f2b((x[i] - mean) * rstd * gma[d] + bta[d]);
  }
}

// ---------------------------------------------------------------------------
extern "C" void kernel_launch(void* const* d_in, const int* in_sizes, int n_in,
                              void* d_out, int out_size, void* d_ws, size_t ws_size,
                              hipStream_t stream) {
  const float* coords     = (const float*)d_in[0];
  const float* features   = (const float*)d_in[1];
  const float* ln1_g      = (const float*)d_in[2];
  const float* ln1_b      = (const float*)d_in[3];
  const float* w_qkv      = (const float*)d_in[4];
  const float* w_attn_out = (const float*)d_in[5];
  const float* b_attn_out = (const float*)d_in[6];
  const float* ga_wq      = (const float*)d_in[7];
  const float* ga_wk      = (const float*)d_in[8];
  const float* ga_wv      = (const float*)d_in[9];
  const float* merge_w    = (const float*)d_in[10];
  const float* merge_b    = (const float*)d_in[11];
  const float* ln2_g      = (const float*)d_in[12];
  const float* ln2_b      = (const float*)d_in[13];
  const float* ff_w1      = (const float*)d_in[14];
  const float* ff_b1      = (const float*)d_in[15];
  const float* ff_w2      = (const float*)d_in[16];
  const float* ff_b2      = (const float*)d_in[17];

  char* ws = (char*)d_ws;
  size_t off = 0;
  auto alloc = [&](size_t bytes) {
    size_t o = off; off += (bytes + 255) & ~(size_t)255; return (void*)(ws + o);
  };
  float*          f_f32  = (float*)alloc((size_t)MROWS * DM * 4);
  float*          f2_f32 = (float*)alloc((size_t)MROWS * DM * 4);
  unsigned short* nf_h   = (unsigned short*)alloc((size_t)MROWS * DM * 2);
  unsigned short* qkv_h  = (unsigned short*)alloc((size_t)MROWS * 3 * DM * 2); // reused as merged f32
  unsigned short* attnh  = (unsigned short*)alloc((size_t)MROWS * DM * 2);
  unsigned short* gq_h   = (unsigned short*)alloc((size_t)MROWS * DM * 2);     // reused as h bf16
  unsigned short* u_h    = (unsigned short*)alloc((size_t)MROWS * DM * 2);
  unsigned short* m_h    = (unsigned short*)alloc((size_t)MROWS * DM * 2);
  unsigned short* ab_h   = (unsigned short*)alloc((size_t)MROWS * 2 * DM * 2); // reused as g1 bf16
  int*            knn    = (int*)alloc((size_t)MROWS * KNN * 4);
  unsigned short* wqkv_h = (unsigned short*)alloc((size_t)DM * 3 * DM * 2);
  unsigned short* wao_h  = (unsigned short*)alloc((size_t)DM * DM * 2);
  unsigned short* gwq_h  = (unsigned short*)alloc((size_t)DM * DM * 2);
  unsigned short* gwkT_h = (unsigned short*)alloc((size_t)DM * DM * 2);
  unsigned short* gwv_h  = (unsigned short*)alloc((size_t)DM * DM * 2);
  unsigned short* mw_h   = (unsigned short*)alloc((size_t)2 * DM * DM * 2);
  unsigned short* fw1_h  = (unsigned short*)alloc((size_t)DM * 2 * DM * 2);
  unsigned short* fw2_h  = (unsigned short*)alloc((size_t)2 * DM * DM * 2);
  float* merged_f32 = (float*)qkv_h;   // alias: qkv dead after attention
  unsigned short* h_h = gq_h;          // alias: gq dead after u GEMM
  unsigned short* g1_h = ab_h;         // alias: ab dead after merge GEMM
  (void)ws_size; (void)n_in; (void)in_sizes; (void)out_size;

  auto cvt = [&](const float* s, unsigned short* d, int n) {
    cvt_kernel<<<(n + 255) / 256, 256, 0, stream>>>(s, d, n);
  };
  cvt(w_qkv, wqkv_h, DM * 3 * DM);
  cvt(w_attn_out, wao_h, DM * DM);
  cvt(ga_wq, gwq_h, DM * DM);
  cvt_t_kernel<<<(DM * DM + 255) / 256, 256, 0, stream>>>(ga_wk, gwkT_h, DM, DM);
  cvt(ga_wv, gwv_h, DM * DM);
  cvt(merge_w, mw_h, 2 * DM * DM);
  cvt(ff_w1, fw1_h, DM * 2 * DM);
  cvt(ff_w2, fw2_h, 2 * DM * DM);

  ln1_kernel<<<dim3(NP, BQ), 128, 0, stream>>>(features, ln1_g, ln1_b, f_f32, nf_h);

  // qkv = nf @ w_qkv  (bf16 out)
  launch_gemm<F_BF16>(stream, nf_h, DM, wqkv_h, 3 * DM, nullptr, nullptr, nullptr, qkv_h,
                      3 * DM, 3 * DM, DM);
  // dense attention
  attn_kernel<<<dim3(NP / 16, NH, BQ), 32, 0, stream>>>(qkv_h, attnh);
  // attn out-proj -> ab[:, 0:D]
  launch_gemm<F_BF16 | F_BIAS>(stream, attnh, DM, wao_h, DM, b_attn_out, nullptr, nullptr,
                               ab_h, 2 * DM, DM, DM);
  // KNN
  knn_kernel<<<dim3(NP, BQ), 32, 0, stream>>>(coords, knn);
  // gq = nf @ ga_wq ; u = gq @ ga_wk^T
  launch_gemm<F_BF16>(stream, nf_h, DM, gwq_h, DM, nullptr, nullptr, nullptr, gq_h, DM, DM, DM);
  launch_gemm<F_BF16>(stream, gq_h, DM, gwkT_h, DM, nullptr, nullptr, nullptr, u_h, DM, DM, DM);
  // m = softmax(nbrs . u) @ nbrs
  geo_kernel<<<MROWS, 128, 0, stream>>>(nf_h, u_h, knn, m_h);
  // geom = m @ ga_wv -> ab[:, D:2D]
  launch_gemm<F_BF16>(stream, m_h, DM, gwv_h, DM, nullptr, nullptr, nullptr, ab_h + DM,
                      2 * DM, DM, DM);
  // merged = ab @ merge_w + merge_b  (f32)
  launch_gemm<F_BIAS>(stream, ab_h, 2 * DM, mw_h, DM, merge_b, nullptr, merged_f32, nullptr,
                      DM, DM, 2 * DM);
  // f2 = f + merged ; h = LN2(f2)
  ln2_kernel<<<MROWS, 128, 0, stream>>>(f_f32, merged_f32, ln2_g, ln2_b, f2_f32, h_h);
  // g1 = gelu(h @ ff_w1 + ff_b1)  (bf16)
  launch_gemm<F_GELU | F_BF16 | F_BIAS>(stream, h_h, DM, fw1_h, 2 * DM, ff_b1, nullptr,
                                        nullptr, g1_h, 2 * DM, 2 * DM, DM);
  // out[b,d,n] = f2 + (g1 @ ff_w2 + ff_b2)   (transposed residual store)
  launch_gemm<F_RESID | F_BIAS>(stream, g1_h, 2 * DM, fw2_h, DM, ff_b2, f2_f32,
                                (float*)d_out, nullptr, DM, DM, 2 * DM);
}
